// TransformerBlock_19086834663774
// MI455X (gfx1250) — matrix operand order repaired and checked by reading, not compiled, not run on hardware
//
#include <hip/hip_runtime.h>
#include <hip/hip_bf16.h>

typedef __attribute__((ext_vector_type(16))) _Float16 v16h;
typedef __attribute__((ext_vector_type(8)))  _Float16 v8h;
typedef __attribute__((ext_vector_type(8)))  float    v8f;

#define BD    2
#define HH    384
#define WW_   384
#define CC    64
#define WIN   8
#define NTOK  64
#define HID   256
#define NW    48
#define SCALE 0.17677669529663687f  /* 1/sqrt(32) */

// ws layout: [0, 75497472) fp32 xattn ; then transposed f16 weights
#define WS_XATTN_FLOATS (BD * HH * WW_ * CC)          // 18,874,368
#define WS_W_BYTEOFF    (WS_XATTN_FLOATS * 4)         // 75,497,472 (16B aligned)
#define WOFF_QKVT 0          // [192][64] f16
#define WOFF_PT   12288      // [64][64]  f16
#define WOFF_W1T  16384      // [256][64] f16
#define WOFF_W2T  32768      // [64][256] f16

// ---------- CDNA5 async global->LDS copy (ASYNCcnt path) ----------
// GVS mode: lds-addr VGPR (flat shared ptr truncates to LDS byte addr),
// per-lane u32 byte offset VGPR, uniform 64-bit SGPR base.
__device__ __forceinline__ void async_g2l_b128(void* lds, const void* gbase, unsigned byte_off) {
  unsigned ldsa = (unsigned)(size_t)lds;
  asm volatile("global_load_async_to_lds_b128 %0, %1, %2"
               :: "v"(ldsa), "v"(byte_off), "s"(gbase)
               : "memory");
}
__device__ __forceinline__ void async_wait0() {
  asm volatile("s_wait_asynccnt 0" ::: "memory");
}

// ---------- WMMA fragment helpers (CDNA5 16x16x32 f16, wave32) ----------
__device__ __forceinline__ v16h load_a(const _Float16* M, int ld, int row0, int k0, int lane) {
  int row = row0 + (lane & 15);
  int p   = lane >> 4;
  const _Float16* s = M + row * ld + k0 + 8 * p;
  v8h lo = *(const v8h*)s;
  v8h hi = *(const v8h*)(s + 16);
  return __builtin_shufflevector(lo, hi, 0,1,2,3,4,5,6,7,8,9,10,11,12,13,14,15);
}
__device__ __forceinline__ v16h load_bt(const _Float16* Mt, int ld, int k0, int col0, int lane) {
  int col = col0 + (lane & 15);
  int p   = lane >> 4;
  const _Float16* s = Mt + col * ld + k0 + 8 * p;
  v8h lo = *(const v8h*)s;
  v8h hi = *(const v8h*)(s + 16);
  return __builtin_shufflevector(lo, hi, 0,1,2,3,4,5,6,7,8,9,10,11,12,13,14,15);
}
__device__ __forceinline__ v8f wmma_f16(v16h a, v16h b, v8f c) {
  return __builtin_amdgcn_wmma_f32_16x16x32_f16(false, a, false, b, (short)0, c, false, false);
}
__device__ __forceinline__ float gelu_exact(float x) {
  return 0.5f * x * (1.0f + erff(x * 0.70710678118654752f));
}
__device__ __forceinline__ float rel_bias_at(const float* relb, int row, int col, int h) {
  int di = (row >> 3) - (col >> 3) + 7;
  int dj = (row & 7)  - (col & 7)  + 7;
  return relb[(di * 15 + dj) * 2 + h];
}

// =====================================================================
// Kernel W: one-time transpose+f16 conversion of all GEMM B-operands.
// =====================================================================
__global__ __launch_bounds__(256)
void weight_prep_kernel(const float* __restrict__ wq, const float* __restrict__ wkv,
                        const float* __restrict__ wproj,
                        const float* __restrict__ w1, const float* __restrict__ w2,
                        _Float16* __restrict__ wsW)
{
  int tid = blockIdx.x * 256 + threadIdx.x;
  int stride = gridDim.x * 256;
  _Float16* Wqkv = wsW + WOFF_QKVT;
  _Float16* Wp   = wsW + WOFF_PT;
  _Float16* W1   = wsW + WOFF_W1T;
  _Float16* W2   = wsW + WOFF_W2T;
  for (int idx = tid; idx < 192 * 64; idx += stride) {
    int o = idx >> 6, in = idx & 63;
    Wqkv[idx] = (_Float16)((o < 64) ? wq[in * 64 + o] : wkv[in * 128 + (o - 64)]);
  }
  for (int idx = tid; idx < 64 * 64; idx += stride) {
    int o = idx >> 6, in = idx & 63;
    Wp[idx] = (_Float16)wproj[in * 64 + o];
  }
  for (int idx = tid; idx < 256 * 64; idx += stride) {
    int o = idx >> 6, in = idx & 63;
    W1[idx] = (_Float16)w1[in * 256 + o];
  }
  for (int idx = tid; idx < 64 * 256; idx += stride) {
    int o = idx >> 8, in = idx & 255;
    W2[idx] = (_Float16)w2[in * 64 + o];
  }
}

// =====================================================================
// Kernel A: LN1 + windowed MHSA + proj + residual  ->  xattn (d_ws)
// 1 block = 1 window (64 tokens x 64 ch), 128 threads = 4 waves.
// =====================================================================
__global__ __launch_bounds__(128)
void win_attn_kernel(const float* __restrict__ x,
                     const float* __restrict__ n1g, const float* __restrict__ n1b,
                     const float* __restrict__ bq,  const float* __restrict__ bkv,
                     const float* __restrict__ relb,
                     const float* __restrict__ bproj,
                     const _Float16* __restrict__ wsW,
                     float* __restrict__ xattn)
{
  extern __shared__ __align__(16) char smem[];
  _Float16* sWqkvT = (_Float16*)(smem);           // [192][64] (transposed)
  _Float16* sWpT   = (_Float16*)(smem + 24576);   // [64][64]  (transposed)
  _Float16* sX     = (_Float16*)(smem + 32768);   // [64][64]  LN'd tokens
  _Float16* sQ     = (_Float16*)(smem + 40960);   // [tok][ch] (pre-scaled)
  _Float16* sK     = (_Float16*)(smem + 49152);   // [tok][ch]
  _Float16* sVt    = (_Float16*)(smem + 57344);   // [ch][tok] (transposed)
  _Float16* sP     = (_Float16*)(smem + 65536);   // [tok][tok] probs
  _Float16* sO     = (_Float16*)(smem + 73728);   // [tok][ch]
  float*    sXf    = (float*)   (smem + 81920);   // [64][64] fp32 staging

  const int tid  = threadIdx.x;
  const int lane = tid & 31;
  const int wv   = tid >> 5;
  const int l    = lane & 15;
  const int p    = lane >> 4;

  const int w   = blockIdx.x;
  const int bi  = w / (NW * NW);
  const int rem = w % (NW * NW);
  const int wh  = rem / NW;
  const int wc  = rem % NW;

  // ---- async stage: window x (fp32) + prepped f16 weights ----
#pragma unroll
  for (int it = 0; it < 8; ++it) {
    int idx = tid + it * 128;            // 0..1023 (16B units)
    int t = idx >> 4, q = idx & 15;
    int i = t >> 3,  j = t & 7;
    unsigned off = (unsigned)((((unsigned)bi * HH * WW_ +
                    (unsigned)(wh * WIN + i) * WW_ + (wc * WIN + j)) * CC) * 4 + q * 16);
    async_g2l_b128(sXf + idx * 4, x, off);
  }
#pragma unroll
  for (int it = 0; it < 12; ++it) {      // 24576 B of WqkvT
    int idx = tid + it * 128;
    async_g2l_b128(sWqkvT + idx * 8, wsW, (unsigned)(WOFF_QKVT * 2 + idx * 16));
  }
#pragma unroll
  for (int it = 0; it < 4; ++it) {       // 8192 B of WprojT
    int idx = tid + it * 128;
    async_g2l_b128(sWpT + idx * 8, wsW, (unsigned)(WOFF_PT * 2 + idx * 16));
  }
  async_wait0();
  __syncthreads();

  // ---- LayerNorm per token ----
  if (tid < 64) {
    const float* r = sXf + tid * 64;
    float m = 0.f;
#pragma unroll 8
    for (int c = 0; c < 64; ++c) m += r[c];
    m *= (1.f / 64.f);
    float v = 0.f;
#pragma unroll 8
    for (int c = 0; c < 64; ++c) { float d = r[c] - m; v += d * d; }
    float rs = rsqrtf(v * (1.f / 64.f) + 1e-5f);
#pragma unroll 8
    for (int c = 0; c < 64; ++c)
      sX[tid * 64 + c] = (_Float16)((r[c] - m) * rs * n1g[c] + n1b[c]);
  }
  __syncthreads();

  // ---- QKV projection: [64x64] @ [64x192] ----
  {
    v16h xa0 = load_a(sX, 64, wv * 16, 0,  lane);
    v16h xa1 = load_a(sX, 64, wv * 16, 32, lane);
#pragma unroll
    for (int nt = 0; nt < 12; ++nt) {
      v8f acc = {};
      acc = wmma_f16(xa0, load_bt(sWqkvT, 64, 0,  nt * 16, lane), acc);
      acc = wmma_f16(xa1, load_bt(sWqkvT, 64, 32, nt * 16, lane), acc);
      int col = nt * 16 + l;
      float bias = (col < 64) ? bq[col] : bkv[col - 64];
#pragma unroll
      for (int r = 0; r < 8; ++r) {
        int row = wv * 16 + r + 8 * p;
        float val = acc[r] + bias;
        if (nt < 4)      sQ[row * 64 + col]          = (_Float16)(val * SCALE);
        else if (nt < 8) sK[row * 64 + (col - 64)]   = (_Float16)val;
        else             sVt[(col - 128) * 64 + row] = (_Float16)val; // transpose V
      }
    }
  }
  __syncthreads();

  // ---- per-head: S = Q.K^T + bias, softmax, O = P.V ----
#pragma unroll
  for (int h = 0; h < 2; ++h) {
    const int hd0 = h * 32;
    v8f s0 = {}, s1 = {}, s2 = {}, s3 = {};
    {
      v16h a = load_a(sQ, 64, wv * 16, hd0, lane);
      s0 = wmma_f16(a, load_bt(sK, 64, hd0,  0, lane), s0);
      s1 = wmma_f16(a, load_bt(sK, 64, hd0, 16, lane), s1);
      s2 = wmma_f16(a, load_bt(sK, 64, hd0, 32, lane), s2);
      s3 = wmma_f16(a, load_bt(sK, 64, hd0, 48, lane), s3);
    }
#pragma unroll
    for (int r = 0; r < 8; ++r) {
      int row = wv * 16 + r + 8 * p;
      s0[r] += rel_bias_at(relb, row,      l, h);
      s1[r] += rel_bias_at(relb, row, 16 + l, h);
      s2[r] += rel_bias_at(relb, row, 32 + l, h);
      s3[r] += rel_bias_at(relb, row, 48 + l, h);
      float mx = fmaxf(fmaxf(s0[r], s1[r]), fmaxf(s2[r], s3[r]));
      mx = fmaxf(mx, __shfl_xor(mx, 1));
      mx = fmaxf(mx, __shfl_xor(mx, 2));
      mx = fmaxf(mx, __shfl_xor(mx, 4));
      mx = fmaxf(mx, __shfl_xor(mx, 8));
      float e0 = __expf(s0[r] - mx), e1 = __expf(s1[r] - mx);
      float e2 = __expf(s2[r] - mx), e3 = __expf(s3[r] - mx);
      float sum = e0 + e1 + e2 + e3;
      sum += __shfl_xor(sum, 1);
      sum += __shfl_xor(sum, 2);
      sum += __shfl_xor(sum, 4);
      sum += __shfl_xor(sum, 8);
      float inv = 1.0f / sum;
      sP[row * 64 +      l] = (_Float16)(e0 * inv);
      sP[row * 64 + 16 + l] = (_Float16)(e1 * inv);
      sP[row * 64 + 32 + l] = (_Float16)(e2 * inv);
      sP[row * 64 + 48 + l] = (_Float16)(e3 * inv);
    }
    // wave-local sP write->read: same-wave DS ordering, no barrier needed
    v16h pa0 = load_a(sP, 64, wv * 16, 0,  lane);
    v16h pa1 = load_a(sP, 64, wv * 16, 32, lane);
#pragma unroll
    for (int nt = 0; nt < 2; ++nt) {
      v8f acc = {};
      acc = wmma_f16(pa0, load_bt(sVt, 64, 0,  hd0 + nt * 16, lane), acc);
      acc = wmma_f16(pa1, load_bt(sVt, 64, 32, hd0 + nt * 16, lane), acc);
      int col = hd0 + nt * 16 + l;
#pragma unroll
      for (int r = 0; r < 8; ++r)
        sO[(wv * 16 + r + 8 * p) * 64 + col] = (_Float16)acc[r];
    }
  }

  // ---- proj + residual (sO rows are wave-local) ----
  {
    v16h oa0 = load_a(sO, 64, wv * 16, 0,  lane);
    v16h oa1 = load_a(sO, 64, wv * 16, 32, lane);
#pragma unroll
    for (int nt = 0; nt < 4; ++nt) {
      v8f acc = {};
      acc = wmma_f16(oa0, load_bt(sWpT, 64, 0,  nt * 16, lane), acc);
      acc = wmma_f16(oa1, load_bt(sWpT, 64, 32, nt * 16, lane), acc);
      int col = nt * 16 + l;
      float bias = bproj[col];
#pragma unroll
      for (int r = 0; r < 8; ++r) {
        int row = wv * 16 + r + 8 * p;
        int i = row >> 3, j = row & 7;
        size_t g = ((size_t)bi * HH * WW_ + (size_t)(wh * WIN + i) * WW_ +
                    (wc * WIN + j)) * CC + col;
        xattn[g] = x[g] + acc[r] + bias;
      }
    }
  }
}

// =====================================================================
// Kernel B: LN2 + LeFF (64->256, gelu, dw3x3, gelu, 256->64) + residual
// 1 block = 8x8 spatial tile (10x10 halo), 128 threads = 4 waves.
// =====================================================================
__global__ __launch_bounds__(128)
void leff_kernel(const float* __restrict__ xattn,
                 const float* __restrict__ n2g, const float* __restrict__ n2b,
                 const float* __restrict__ b1,
                 const float* __restrict__ dwk, const float* __restrict__ dwb,
                 const float* __restrict__ b2,
                 const _Float16* __restrict__ wsW,
                 float* __restrict__ out)
{
  extern __shared__ __align__(16) char smem[];
  _Float16* sW1T = (_Float16*)(smem);             // [256][64]  (transposed)
  _Float16* sW2T = (_Float16*)(smem +  32768);    // [64][256]  (transposed)
  _Float16* sY   = (_Float16*)(smem +  65536);    // [112][64]  LN'd halo (pad)
  _Float16* sH   = (_Float16*)(smem +  79872);    // [112][256] hidden
  _Float16* sG   = (_Float16*)(smem + 137216);    // [64][256]  post-conv
  float*    sHf  = (float*)sH;                    // fp32 halo staging (25.6KB, reused)

  const int tid  = threadIdx.x;
  const int lane = tid & 31;
  const int wv   = tid >> 5;
  const int l    = lane & 15;
  const int p    = lane >> 4;

  const int w   = blockIdx.x;
  const int bi  = w / (NW * NW);
  const int rem = w % (NW * NW);
  const int ty  = rem / NW, tx = rem % NW;
  const int oy  = ty * 8 - 1, ox = tx * 8 - 1;    // halo origin

  // ---- async stage: prepped weights + in-bounds halo rows (fp32) ----
#pragma unroll
  for (int it = 0; it < 16; ++it) {      // 32768 B of W1T
    int idx = tid + it * 128;
    async_g2l_b128(sW1T + idx * 8, wsW, (unsigned)(WOFF_W1T * 2 + idx * 16));
  }
#pragma unroll
  for (int it = 0; it < 16; ++it) {      // 32768 B of W2T
    int idx = tid + it * 128;
    async_g2l_b128(sW2T + idx * 8, wsW, (unsigned)(WOFF_W2T * 2 + idx * 16));
  }
  for (int idx = tid; idx < 100 * 16; idx += 128) {   // 100 rows x 256B
    int t = idx >> 4, q = idx & 15;
    int hy = oy + t / 10, hx = ox + t % 10;
    if (hy >= 0 && hy < HH && hx >= 0 && hx < WW_) {
      unsigned off = (unsigned)((((unsigned)bi * HH * WW_ +
                      (unsigned)hy * WW_ + hx) * CC) * 4 + q * 16);
      async_g2l_b128(sHf + idx * 4, xattn, off);
    }
  }
  async_wait0();
  __syncthreads();

  // ---- LN2 over halo tokens (zeros for OOB + pad rows) ----
  for (int t = tid; t < 112; t += 128) {
    int hy = oy + t / 10, hx = ox + t % 10;
    bool ok = (t < 100) && hy >= 0 && hy < HH && hx >= 0 && hx < WW_;
    if (ok) {
      const float* r = sHf + t * 64;
      float m = 0.f;
#pragma unroll 8
      for (int c = 0; c < 64; ++c) m += r[c];
      m *= (1.f / 64.f);
      float v = 0.f;
#pragma unroll 8
      for (int c = 0; c < 64; ++c) { float d = r[c] - m; v += d * d; }
      float rs = rsqrtf(v * (1.f / 64.f) + 1e-5f);
#pragma unroll 8
      for (int c = 0; c < 64; ++c)
        sY[t * 64 + c] = (_Float16)((r[c] - m) * rs * n2g[c] + n2b[c]);
    } else {
#pragma unroll 8
      for (int c = 0; c < 64; ++c) sY[t * 64 + c] = (_Float16)0.f;
    }
  }
  __syncthreads();   // sY ready; sHf reads done (sH overwritten below)

  // ---- GEMM1: h = gelu(y @ w1 + b1)  (7 M-tiles over 4 waves) ----
  for (int mt = wv; mt < 7; mt += 4) {
    v16h ya0 = load_a(sY, 64, mt * 16, 0,  lane);
    v16h ya1 = load_a(sY, 64, mt * 16, 32, lane);
#pragma unroll
    for (int nt = 0; nt < 16; ++nt) {
      v8f acc = {};
      acc = wmma_f16(ya0, load_bt(sW1T, 64, 0,  nt * 16, lane), acc);
      acc = wmma_f16(ya1, load_bt(sW1T, 64, 32, nt * 16, lane), acc);
      int col = nt * 16 + l;
      float bias = b1[col];
#pragma unroll
      for (int r = 0; r < 8; ++r) {
        int row = mt * 16 + r + 8 * p;
        sH[row * 256 + col] = (_Float16)gelu_exact(acc[r] + bias);
      }
    }
  }
  __syncthreads();

  // ---- depthwise 3x3 (zero pad at image edges) + bias + gelu ----
  for (int idx = tid; idx < 64 * 256; idx += 128) {
    int t = idx >> 8, c = idx & 255;
    int i = t >> 3,  j = t & 7;
    float acc = dwb[c];
#pragma unroll
    for (int di = 0; di < 3; ++di)
#pragma unroll
      for (int dj = 0; dj < 3; ++dj) {
        int hy = oy + i + di, hx = ox + j + dj;
        if (hy >= 0 && hy < HH && hx >= 0 && hx < WW_)
          acc += dwk[c * 9 + di * 3 + dj] * (float)sH[((i + di) * 10 + (j + dj)) * 256 + c];
      }
    sG[t * 256 + c] = (_Float16)gelu_exact(acc);
  }
  __syncthreads();

  // ---- GEMM2 + residual ----
  {
    v16h ga[8];
#pragma unroll
    for (int kt = 0; kt < 8; ++kt) ga[kt] = load_a(sG, 256, wv * 16, kt * 32, lane);
#pragma unroll
    for (int nt = 0; nt < 4; ++nt) {
      v8f acc = {};
#pragma unroll
      for (int kt = 0; kt < 8; ++kt)
        acc = wmma_f16(ga[kt], load_bt(sW2T, 256, kt * 32, nt * 16, lane), acc);
      int col = nt * 16 + l;
      float bias = b2[col];
#pragma unroll
      for (int r = 0; r < 8; ++r) {
        int row = wv * 16 + r + 8 * p;
        int i = row >> 3, j = row & 7;
        size_t g = ((size_t)bi * HH * WW_ + (size_t)(ty * 8 + i) * WW_ +
                    (tx * 8 + j)) * CC + col;
        out[g] = xattn[g] + acc[r] + bias;
      }
    }
  }
}

extern "C" void kernel_launch(void* const* d_in, const int* in_sizes, int n_in,
                              void* d_out, int out_size, void* d_ws, size_t ws_size,
                              hipStream_t stream)
{
  (void)in_sizes; (void)n_in; (void)out_size; (void)ws_size;
  const float* x     = (const float*)d_in[0];
  const float* n1g   = (const float*)d_in[1];
  const float* n1b   = (const float*)d_in[2];
  const float* wq    = (const float*)d_in[3];
  const float* bq    = (const float*)d_in[4];
  const float* wkv   = (const float*)d_in[5];
  const float* bkv   = (const float*)d_in[6];
  const float* relb  = (const float*)d_in[7];
  const float* wproj = (const float*)d_in[8];
  const float* bproj = (const float*)d_in[9];
  const float* n2g   = (const float*)d_in[10];
  const float* n2b   = (const float*)d_in[11];
  const float* w1    = (const float*)d_in[12];
  const float* b1    = (const float*)d_in[13];
  const float* dwk   = (const float*)d_in[14];
  const float* dwb   = (const float*)d_in[15];
  const float* w2    = (const float*)d_in[16];
  const float* b2    = (const float*)d_in[17];
  float*    out   = (float*)d_out;
  float*    xattn = (float*)d_ws;                               // 75.5 MB
  _Float16* wsW   = (_Float16*)((char*)d_ws + WS_W_BYTEOFF);    // +96 KB f16 weights

  dim3 grid(BD * NW * NW);       // 4608 blocks
  dim3 blk(128);                 // 4 wave32

  weight_prep_kernel<<<dim3(32), dim3(256), 0, stream>>>(wq, wkv, wproj, w1, w2, wsW);
  win_attn_kernel<<<grid, blk, 98304, stream>>>(
      x, n1g, n1b, bq, bkv, relb, bproj, wsW, xattn);
  leff_kernel<<<grid, blk, 169984, stream>>>(
      xattn, n2g, n2b, b1, dwk, dwb, b2, wsW, out);
}
